// crossAttentionProcessorAmplify_55379308315030
// MI455X (gfx1250) — compile-verified, matrix operand order checked by reading, not run on hardware
//
#include <hip/hip_runtime.h>
#include <hip/hip_bf16.h>

typedef __bf16 bf16;
typedef __attribute__((ext_vector_type(16))) __bf16 v16bf;
typedef __attribute__((ext_vector_type(8)))  float  v8f;

#define NB    4        // batch
#define SQ    16384    // query tokens
#define CC    320      // inner dim
#define SKV   77       // kv tokens
#define CX    768      // cross dim
#define NH    8        // heads
#define DH    40       // head dim
#define DPAD  64       // head dim padded for WMMA (2 k-steps of 32)
#define KVPAD 96       // kv padded for WMMA (3 k-steps of 32, 6 n-tiles of 16)
#define MROWS (NB*SQ)  // 65536 flattened rows

// ---- WMMA fragment helpers ------------------------------------------------
// 16-bit A-matrix 16x32 layout (ISA 7.12.2): lane&15 = row M; lanes<16 hold
// K {0..7, 16..23}, lanes>=16 hold K {8..15, 24..31}. B uses the mirrored
// layout (lane&15 = col N, same K split), so one helper serves both given a
// pointer to the start of the row (A) / column (B) with contiguous K.
__device__ __forceinline__ v16bf fragK_bf16(const bf16* row, int kbase, bool hi) {
  v16bf f;
  int o = kbase + (hi ? 8 : 0);
#pragma unroll
  for (int j = 0; j < 8; ++j) { f[j] = row[o + j]; f[j + 8] = row[o + 16 + j]; }
  return f;
}
__device__ __forceinline__ v16bf fragK_f32(const float* row, int kbase, bool hi) {
  v16bf f;
  int o = kbase + (hi ? 8 : 0);
#pragma unroll
  for (int j = 0; j < 8; ++j) { f[j] = (bf16)row[o + j]; f[j + 8] = (bf16)row[o + 16 + j]; }
  return f;
}
__device__ __forceinline__ v8f wmma_bf16(v16bf a, v16bf b, v8f c) {
  return __builtin_amdgcn_wmma_f32_16x16x32_bf16(false, a, false, b, (short)0, c, false, false);
}

// ---- Kernel 1: weight transpose to bf16 [N][K] -----------------------------
__global__ __launch_bounds__(256) void prep_weights(const float* __restrict__ Wq,
                                                    const float* __restrict__ Wo,
                                                    bf16* __restrict__ Wq_t,
                                                    bf16* __restrict__ Wo_t) {
  int i = blockIdx.x * blockDim.x + threadIdx.x;
  if (i >= CC * CC) return;
  int n = i / CC, k = i % CC;
  Wq_t[n * CC + k] = (bf16)Wq[k * CC + n];
  Wo_t[n * CC + k] = (bf16)Wo[k * CC + n];
}

// ---- Kernel 2: K/V projection + amplify, padded WMMA layouts ---------------
// Kt: [b][h][kv (pad 96)][d (pad 64)]  (B-fragments of scores: contiguous d)
// Vt: [b][h][d  (pad 64)][kv (pad 96)] (B-fragments of P*V:   contiguous kv)
__global__ __launch_bounds__(256) void prep_kv(const float* __restrict__ enc,
                                               const float* __restrict__ Wk,
                                               const float* __restrict__ Wv,
                                               const int*   __restrict__ tstep,
                                               bf16* __restrict__ Kt,
                                               bf16* __restrict__ Vt) {
  int i = blockIdx.x * blockDim.x + threadIdx.x;
  if (i >= NB * SKV * CC) return;
  int c  = i % CC;
  int kv = (i / CC) % SKV;
  int b  = i / (CC * SKV);
  const float* e = enc + (size_t)(b * SKV + kv) * CX;
  float kacc = 0.f, vacc = 0.f;
  for (int k = 0; k < CX; ++k) {
    float x = e[k];
    kacc = fmaf(x, Wk[k * CC + c], kacc);
    vacc = fmaf(x, Wv[k * CC + c], vacc);
  }
  float amp = fmaxf(1.0f, (float)(tstep[0] - 1) * 0.02f);
  if ((b == 1 || b == 3) && (kv == 2 || kv == 5)) vacc *= amp;
  int h = c / DH, d = c % DH;
  Kt[(((size_t)(b * NH + h) * KVPAD) + kv) * DPAD + d] = (bf16)kacc;
  Vt[(((size_t)(b * NH + h) * DPAD) + d) * KVPAD + kv] = (bf16)vacc;
}

// ---- Kernel 3: Q = hs @ Wq  (bf16 WMMA, 16x64 register block per wave) -----
// Output layout: Qpad [row=b*SQ+s][h][64], d 40..63 pre-zeroed by memset.
// 4 independent accumulators per wave: A fragment loaded once per k-step and
// reused 4x; independent wmmas remove the RAW hazard NOP chains.
__global__ __launch_bounds__(256) void gemm_q(const float* __restrict__ hs,
                                              const bf16*  __restrict__ Wq_t,
                                              bf16* __restrict__ Qpad) {
  int wave = (blockIdx.x * blockDim.x + threadIdx.x) >> 5;
  int lane = threadIdx.x & 31;
  const int TG = CC / 64;          // 5 column groups of 64
  int tm = wave / TG, tg = wave % TG;
  bool hi = lane >= 16;
  int nlo = lane & 15;
  const float* Arow = hs + (size_t)(tm * 16 + nlo) * CC;
  const bf16* B0 = Wq_t + (size_t)(tg * 64 + 0 * 16 + nlo) * CC;
  const bf16* B1 = Wq_t + (size_t)(tg * 64 + 1 * 16 + nlo) * CC;
  const bf16* B2 = Wq_t + (size_t)(tg * 64 + 2 * 16 + nlo) * CC;
  const bf16* B3 = Wq_t + (size_t)(tg * 64 + 3 * 16 + nlo) * CC;
  v8f c0 = {}, c1 = {}, c2 = {}, c3 = {};
#pragma unroll
  for (int ks = 0; ks < CC / 32; ++ks) {
    v16bf a = fragK_f32(Arow, ks * 32, hi);
    c0 = wmma_bf16(a, fragK_bf16(B0, ks * 32, hi), c0);
    c1 = wmma_bf16(a, fragK_bf16(B1, ks * 32, hi), c1);
    c2 = wmma_bf16(a, fragK_bf16(B2, ks * 32, hi), c2);
    c3 = wmma_bf16(a, fragK_bf16(B3, ks * 32, hi), c3);
  }
  // C/D layout: VGPR j -> M = j (+8 for upper lanes), lane&15 -> N
  v8f acc[4] = {c0, c1, c2, c3};
  int mb = tm * 16 + (hi ? 8 : 0);
#pragma unroll
  for (int sub = 0; sub < 4; ++sub) {
    int c = tg * 64 + sub * 16 + nlo;
    int h = c / DH, d = c % DH;
#pragma unroll
    for (int j = 0; j < 8; ++j) {
      Qpad[((size_t)(mb + j) * NH + h) * DPAD + d] = (bf16)acc[sub][j];
    }
  }
}

// ---- Kernel 4: fused attention per (b, h, 16-query tile) -------------------
__global__ __launch_bounds__(128) void attn(const bf16* __restrict__ Qpad,
                                            const bf16* __restrict__ Kt,
                                            const bf16* __restrict__ Vt,
                                            bf16* __restrict__ attnO) {
  __shared__ float sc[4][16][KVPAD];   // scores, per wave
  __shared__ bf16  pb[4][16][KVPAD];   // probs (bf16), per wave
  int wv   = threadIdx.x >> 5;
  int lane = threadIdx.x & 31;
  int w    = blockIdx.x * 4 + wv;      // total waves = 4*8*1024 = 32768
  int qt = w & 1023;
  int h  = (w >> 10) & 7;
  int b  = w >> 13;
  bool hi = lane >= 16;
  int nlo = lane & 15;
  int q0 = qt * 16;

  // ---- scores = Q @ K^T (scaled in softmax) : 2 k-steps x 6 kv-tiles ----
  const bf16* Qrow = Qpad + ((size_t)(b * SQ + q0 + nlo) * NH + h) * DPAD;
  v8f s0 = {}, s1 = {}, s2 = {}, s3 = {}, s4 = {}, s5 = {};
#pragma unroll
  for (int ks = 0; ks < 2; ++ks) {
    v16bf a = fragK_bf16(Qrow, ks * 32, hi);
    const bf16* Kb = Kt + ((size_t)(b * NH + h) * KVPAD + nlo) * DPAD;
    s0 = wmma_bf16(a, fragK_bf16(Kb + (size_t)(0 * 16) * DPAD, ks * 32, hi), s0);
    s1 = wmma_bf16(a, fragK_bf16(Kb + (size_t)(1 * 16) * DPAD, ks * 32, hi), s1);
    s2 = wmma_bf16(a, fragK_bf16(Kb + (size_t)(2 * 16) * DPAD, ks * 32, hi), s2);
    s3 = wmma_bf16(a, fragK_bf16(Kb + (size_t)(3 * 16) * DPAD, ks * 32, hi), s3);
    s4 = wmma_bf16(a, fragK_bf16(Kb + (size_t)(4 * 16) * DPAD, ks * 32, hi), s4);
    s5 = wmma_bf16(a, fragK_bf16(Kb + (size_t)(5 * 16) * DPAD, ks * 32, hi), s5);
  }
  int mb = hi ? 8 : 0;
#pragma unroll
  for (int j = 0; j < 8; ++j) {
    sc[wv][mb + j][0 * 16 + nlo] = s0[j];
    sc[wv][mb + j][1 * 16 + nlo] = s1[j];
    sc[wv][mb + j][2 * 16 + nlo] = s2[j];
    sc[wv][mb + j][3 * 16 + nlo] = s3[j];
    sc[wv][mb + j][4 * 16 + nlo] = s4[j];
    sc[wv][mb + j][5 * 16 + nlo] = s5[j];
  }
  asm volatile("s_wait_dscnt 0" ::: "memory");

  // ---- softmax over kv (mask kv>=77), lanes 0..15 handle one row each ----
  if (!hi) {
    const float scale = 0.158113883008419f;  // 1/sqrt(40)
    int r = nlo;
    float mx = -3.0e38f;
    for (int c = 0; c < SKV; ++c) {
      float x = sc[wv][r][c] * scale;
      sc[wv][r][c] = x;
      mx = fmaxf(mx, x);
    }
    float sum = 0.f;
    for (int c = 0; c < SKV; ++c) {
      float e = __expf(sc[wv][r][c] - mx);
      sum += e;
      sc[wv][r][c] = e;
    }
    float inv = 1.0f / sum;
    for (int c = 0; c < SKV; ++c)     pb[wv][r][c] = (bf16)(sc[wv][r][c] * inv);
    for (int c = SKV; c < KVPAD; ++c) pb[wv][r][c] = (bf16)0.0f;
  }
  asm volatile("s_wait_dscnt 0" ::: "memory");

  // ---- out = P @ V : 3 k-steps x 4 d-tiles ----
  const bf16* Prow = &pb[wv][nlo][0];
  v8f o0 = {}, o1 = {}, o2 = {}, o3 = {};
#pragma unroll
  for (int ks = 0; ks < 3; ++ks) {
    v16bf a = fragK_bf16(Prow, ks * 32, hi);
    const bf16* Vb = Vt + ((size_t)(b * NH + h) * DPAD + nlo) * KVPAD;
    o0 = wmma_bf16(a, fragK_bf16(Vb + (size_t)(0 * 16) * KVPAD, ks * 32, hi), o0);
    o1 = wmma_bf16(a, fragK_bf16(Vb + (size_t)(1 * 16) * KVPAD, ks * 32, hi), o1);
    o2 = wmma_bf16(a, fragK_bf16(Vb + (size_t)(2 * 16) * KVPAD, ks * 32, hi), o2);
    o3 = wmma_bf16(a, fragK_bf16(Vb + (size_t)(3 * 16) * KVPAD, ks * 32, hi), o3);
  }
  v8f oacc[4] = {o0, o1, o2, o3};
#pragma unroll
  for (int nt = 0; nt < 4; ++nt) {
    int d = nt * 16 + nlo;
    if (d < DH) {
#pragma unroll
      for (int j = 0; j < 8; ++j) {
        int r = q0 + mb + j;
        attnO[((size_t)(b * SQ + r)) * CC + h * DH + d] = (bf16)oacc[nt][j];
      }
    }
  }
}

// ---- Kernel 5: out = attnO @ Wo + bo + residual (16x64 block per wave) -----
__global__ __launch_bounds__(256) void gemm_out(const bf16* __restrict__ attnO,
                                                const bf16* __restrict__ Wo_t,
                                                const float* __restrict__ bo,
                                                const float* __restrict__ hs,
                                                float* __restrict__ out) {
  int wave = (blockIdx.x * blockDim.x + threadIdx.x) >> 5;
  int lane = threadIdx.x & 31;
  const int TG = CC / 64;          // 5 column groups of 64
  int tm = wave / TG, tg = wave % TG;
  bool hi = lane >= 16;
  int nlo = lane & 15;
  const bf16* Arow = attnO + (size_t)(tm * 16 + nlo) * CC;
  const bf16* B0 = Wo_t + (size_t)(tg * 64 + 0 * 16 + nlo) * CC;
  const bf16* B1 = Wo_t + (size_t)(tg * 64 + 1 * 16 + nlo) * CC;
  const bf16* B2 = Wo_t + (size_t)(tg * 64 + 2 * 16 + nlo) * CC;
  const bf16* B3 = Wo_t + (size_t)(tg * 64 + 3 * 16 + nlo) * CC;
  v8f c0 = {}, c1 = {}, c2 = {}, c3 = {};
#pragma unroll
  for (int ks = 0; ks < CC / 32; ++ks) {
    v16bf a = fragK_bf16(Arow, ks * 32, hi);
    c0 = wmma_bf16(a, fragK_bf16(B0, ks * 32, hi), c0);
    c1 = wmma_bf16(a, fragK_bf16(B1, ks * 32, hi), c1);
    c2 = wmma_bf16(a, fragK_bf16(B2, ks * 32, hi), c2);
    c3 = wmma_bf16(a, fragK_bf16(B3, ks * 32, hi), c3);
  }
  v8f acc[4] = {c0, c1, c2, c3};
  int mb = tm * 16 + (hi ? 8 : 0);
#pragma unroll
  for (int sub = 0; sub < 4; ++sub) {
    int n = tg * 64 + sub * 16 + nlo;
    float bias = bo[n];
#pragma unroll
    for (int j = 0; j < 8; ++j) {
      size_t idx = (size_t)(mb + j) * CC + n;
      out[idx] = acc[sub][j] + bias + hs[idx];
    }
  }
}

// ---- Host side -------------------------------------------------------------
extern "C" void kernel_launch(void* const* d_in, const int* in_sizes, int n_in,
                              void* d_out, int out_size, void* d_ws, size_t ws_size,
                              hipStream_t stream) {
  const float* hs  = (const float*)d_in[0];
  const float* enc = (const float*)d_in[1];
  const float* Wq  = (const float*)d_in[2];
  const float* Wk  = (const float*)d_in[3];
  const float* Wv  = (const float*)d_in[4];
  const float* Wo  = (const float*)d_in[5];
  const float* bo  = (const float*)d_in[6];
  const int*   ts  = (const int*)d_in[7];
  float* out = (float*)d_out;

  size_t off = 0;
  auto carve = [&](size_t bytes) -> void* {
    void* p = (char*)d_ws + off;
    off += (bytes + 255) & ~(size_t)255;
    return p;
  };
  size_t qpad_b = (size_t)MROWS * NH * DPAD * sizeof(bf16);        // 64 MB
  size_t attn_b = (size_t)MROWS * CC * sizeof(bf16);               // 40 MB
  size_t kt_b   = (size_t)NB * NH * KVPAD * DPAD * sizeof(bf16);
  size_t wt_b   = (size_t)CC * CC * sizeof(bf16);
  bf16* Qpad  = (bf16*)carve(qpad_b);
  bf16* attnO = (bf16*)carve(attn_b);
  bf16* Kt    = (bf16*)carve(kt_b);
  bf16* Vt    = (bf16*)carve(kt_b);
  bf16* Wq_t  = (bf16*)carve(wt_b);
  bf16* Wo_t  = (bf16*)carve(wt_b);

  // zero padded regions (deterministic each call; memset is graph-capturable)
  hipMemsetAsync(Qpad, 0, qpad_b, stream);
  hipMemsetAsync(Kt, 0, kt_b, stream);
  hipMemsetAsync(Vt, 0, kt_b, stream);

  prep_weights<<<(CC * CC + 255) / 256, 256, 0, stream>>>(Wq, Wo, Wq_t, Wo_t);
  prep_kv<<<(NB * SKV * CC + 255) / 256, 256, 0, stream>>>(enc, Wk, Wv, ts, Kt, Vt);

  // Q projection: 4096 row-tiles x 5 col-groups = 20480 waves / 8 per block
  gemm_q<<<(MROWS / 16) * (CC / 64) / 8, 256, 0, stream>>>(hs, Wq_t, Qpad);

  // attention: 4*8*1024 = 32768 waves / 4 per block
  attn<<<NB * NH * (SQ / 16) / 4, 128, 0, stream>>>(Qpad, Kt, Vt, attnO);

  // output projection + bias + residual
  gemm_out<<<(MROWS / 16) * (CC / 64) / 8, 256, 0, stream>>>(attnO, Wo_t, bo, hs, out);
}
